// Decoder_1898375545208
// MI455X (gfx1250) — matrix operand order repaired and checked by reading, not compiled, not run on hardware
//
#include <hip/hip_runtime.h>
#include <hip/hip_bf16.h>

// ---------------------------------------------------------------------------
// out[e,c] = relu( u[e] . (M_c . v[e]) ),  M_c = sum_k Wc[k,c] * W_k
// Phase 0: build M_c (bf16), convert user/item tables to bf16.
// Phase 1: tv[i,c,:] = M_c @ item[i]  via v_wmma_f32_16x16x32_bf16.
// Phase 2: per-edge gather + 5 dot products (wave32 per edge, L2-resident).
// ---------------------------------------------------------------------------

typedef __bf16 bf16;
typedef __attribute__((ext_vector_type(16))) __bf16 v16bf;
typedef __attribute__((ext_vector_type(8)))  __bf16 v8bf;
typedef __attribute__((ext_vector_type(4)))  __bf16 v4bf;
typedef __attribute__((ext_vector_type(8)))  float  v8f;

#define N_USERS 100000
#define N_ITEMS 50000
#define DIM     128
#define NEDGE   1000000
#define KBASES  8
#define NC      5

// ---------------------- Phase 0a: fold classifier into bases ----------------
__global__ __launch_bounds__(256)
void k_build_M(const float* __restrict__ W,      // [K,128,128]
               const float* __restrict__ wc,     // [K,5]
               bf16* __restrict__ M)             // [5,128,128] bf16
{
    int idx = blockIdx.x * blockDim.x + threadIdx.x;   // 0..81919
    if (idx >= NC * DIM * DIM) return;
    int c  = idx / (DIM * DIM);
    int df = idx % (DIM * DIM);
    float acc = 0.0f;
#pragma unroll
    for (int k = 0; k < KBASES; ++k)
        acc = fmaf(wc[k * NC + c], W[k * DIM * DIM + df], acc);
    M[idx] = (bf16)acc;
}

// ---------------------- Phase 0b: f32 -> bf16 table convert -----------------
__global__ __launch_bounds__(256)
void k_f32_to_bf16(const float* __restrict__ src, bf16* __restrict__ dst, int n4)
{
    int i = blockIdx.x * blockDim.x + threadIdx.x;     // one float4 per thread
    if (i >= n4) return;
    float4 v = ((const float4*)src)[i];
    v4bf o;
    o[0] = (bf16)v.x; o[1] = (bf16)v.y; o[2] = (bf16)v.z; o[3] = (bf16)v.w;
    ((v4bf*)dst)[i] = o;
}

// ---------------------- Phase 1: tv = items @ M_c^T via WMMA ----------------
// One wave handles a 16-row item tile x all 128 output cols for one class c.
// K=128 -> 4 steps of 32.  8 N-tiles of 16.  => 32 WMMAs / wave.
__global__ __launch_bounds__(256)
void k_tv_wmma(const bf16* __restrict__ items,   // [N_ITEMS,128] bf16
               const bf16* __restrict__ M,       // [5,128,128]  bf16 (B matrix: M_c[d,f])
               bf16* __restrict__ tv)            // [N_ITEMS,5,128] bf16
{
    const int lane = threadIdx.x & 31;
    const int wave = threadIdx.x >> 5;
    const int task = blockIdx.x * 8 + wave;            // 8 waves / block
    const int NTASK = (N_ITEMS / 16) * NC;             // 3125 * 5 = 15625
    if (task >= NTASK) return;                         // wave-uniform exit

    const int c        = task % NC;
    const int row_base = (task / NC) * 16;

    const bf16* Mc = M + (size_t)c * DIM * DIM;

    // A 16x32 layout (matches B): lanes 0-15 row=lane, elems 0..15 -> K 0..15;
    // lanes 16-31 same rows, K 16..31.
    const int arow = lane & 15;
    const int ahk  = (lane >> 4) * 16;                 // 0 or 16
    // B 32x16 layout: col = lane&15; lanes 0-15 hold K 0..15, lanes 16-31 K 16..31.
    const int bhk  = (lane >> 4) * 16;                 // 0 or 16
    const int ncol = lane & 15;

    v8f zero;
#pragma unroll
    for (int i = 0; i < 8; ++i) zero[i] = 0.0f;
    v8f acc[8];
#pragma unroll
    for (int t = 0; t < 8; ++t) acc[t] = zero;

    const bf16* arowp = items + (size_t)(row_base + arow) * DIM;

#pragma unroll
    for (int ks = 0; ks < 4; ++ks) {
        const int k0 = ks * 32;
        // A fragment: 16 contiguous bf16 per lane (two 16B chunks)
        v8bf alo = *(const v8bf*)(arowp + k0 + ahk);
        v8bf ahi = *(const v8bf*)(arowp + k0 + ahk + 8);
        v16bf afrag;
#pragma unroll
        for (int i = 0; i < 8; ++i) { afrag[i] = alo[i]; afrag[8 + i] = ahi[i]; }

#pragma unroll
        for (int nt = 0; nt < 8; ++nt) {
            // B fragment: 16 contiguous bf16 of M_c[row n, k0+bhk ..+15] (32B aligned)
            v16bf bfrag = *(const v16bf*)(Mc + (size_t)(nt * 16 + ncol) * DIM + k0 + bhk);
            acc[nt] = __builtin_amdgcn_wmma_f32_16x16x32_bf16(
                false, afrag, false, bfrag, (short)0, acc[nt], false, false);
        }
    }

    // C/D layout: VGPR r, lanes 0-15 -> M=r, lanes 16-31 -> M=r+8; col = lane&15.
#pragma unroll
    for (int nt = 0; nt < 8; ++nt) {
#pragma unroll
        for (int r = 0; r < 8; ++r) {
            const int m   = r + ((lane >> 4) << 3);
            const int col = nt * 16 + ncol;
            tv[(size_t)(row_base + m) * (NC * DIM) + c * DIM + col] = (bf16)acc[nt][r];
        }
    }
}

// ---------------------- Phase 2: edge gather + 5 dots -----------------------
// One wave32 per edge. Lane handles dims [4*lane, 4*lane+4).
__global__ __launch_bounds__(256)
void k_edges(const bf16* __restrict__ ub,        // [N_USERS,128] bf16
             const bf16* __restrict__ tv,        // [N_ITEMS,5,128] bf16
             const int*  __restrict__ uidx,
             const int*  __restrict__ iidx,
             float* __restrict__ out)            // [E,5] f32
{
    const int lane = threadIdx.x & 31;
    const int e    = (blockIdx.x * blockDim.x + threadIdx.x) >> 5;
    if (e >= NEDGE) return;                      // wave-uniform exit

    const int ui = uidx[e];
    const int ii = iidx[e];

    const bf16* up = ub + (size_t)ui * DIM + lane * 4;
    const bf16* tp = tv + (size_t)ii * (NC * DIM) + lane * 4;
    __builtin_prefetch(tp, 0, 1);                // global_prefetch of tv row

    v4bf uv = *(const v4bf*)up;
    const float u0 = (float)uv[0], u1 = (float)uv[1];
    const float u2 = (float)uv[2], u3 = (float)uv[3];

    float s[NC];
#pragma unroll
    for (int c = 0; c < NC; ++c) {
        v4bf t = *(const v4bf*)(tp + c * DIM);
        float a = u0 * (float)t[0];
        a = fmaf(u1, (float)t[1], a);
        a = fmaf(u2, (float)t[2], a);
        a = fmaf(u3, (float)t[3], a);
        s[c] = a;
    }
    // wave32 butterfly reduction
#pragma unroll
    for (int c = 0; c < NC; ++c) {
        float v = s[c];
        v += __shfl_xor(v, 1);
        v += __shfl_xor(v, 2);
        v += __shfl_xor(v, 4);
        v += __shfl_xor(v, 8);
        v += __shfl_xor(v, 16);
        s[c] = v;
    }
    const float r = (lane == 0) ? s[0] :
                    (lane == 1) ? s[1] :
                    (lane == 2) ? s[2] :
                    (lane == 3) ? s[3] : s[4];
    if (lane < NC) out[(size_t)e * NC + lane] = fmaxf(r, 0.0f);
}

// ---------------------------------------------------------------------------
extern "C" void kernel_launch(void* const* d_in, const int* in_sizes, int n_in,
                              void* d_out, int out_size, void* d_ws, size_t ws_size,
                              hipStream_t stream)
{
    (void)in_sizes; (void)n_in; (void)out_size; (void)ws_size;

    const float* user_inputs  = (const float*)d_in[0];   // [100000,128]
    const float* item_inputs  = (const float*)d_in[1];   // [50000,128]
    const int*   user_indices = (const int*)  d_in[2];   // [1e6]
    const int*   item_indices = (const int*)  d_in[3];   // [1e6]
    const float* weight       = (const float*)d_in[4];   // [8,128,128]
    const float* wclass       = (const float*)d_in[5];   // [8,5]
    float*       out          = (float*)d_out;           // [1e6,5]

    // Workspace layout (all offsets multiple of 256B); total ~102.6 MB.
    char* ws = (char*)d_ws;
    const size_t OFF_M     = 0;                              //   163,840 B
    const size_t OFF_ITEMS = OFF_M     + 163840;             // 12,800,000 B
    const size_t OFF_USERS = OFF_ITEMS + 12800000;           // 25,600,000 B
    const size_t OFF_TV    = OFF_USERS + 25600000;           // 64,000,000 B
    bf16* Mb  = (bf16*)(ws + OFF_M);
    bf16* itb = (bf16*)(ws + OFF_ITEMS);
    bf16* ubf = (bf16*)(ws + OFF_USERS);
    bf16* tvb = (bf16*)(ws + OFF_TV);

    // Phase 0: fold classifier, convert tables to bf16.
    k_build_M<<<(NC * DIM * DIM + 255) / 256, 256, 0, stream>>>(weight, wclass, Mb);
    k_f32_to_bf16<<<(N_ITEMS * DIM / 4 + 255) / 256, 256, 0, stream>>>(item_inputs, itb, N_ITEMS * DIM / 4);
    k_f32_to_bf16<<<(N_USERS * DIM / 4 + 255) / 256, 256, 0, stream>>>(user_inputs, ubf, N_USERS * DIM / 4);

    // Phase 1: tv[i,c,:] = M_c @ item[i]  (WMMA bf16, f32 accumulate)
    const int ntask = (N_ITEMS / 16) * NC;                   // 15625 wave tasks
    k_tv_wmma<<<(ntask + 7) / 8, 256, 0, stream>>>(itb, Mb, tvb);

    // Phase 2: per-edge gather + dots (wave per edge: 1e6 waves)
    k_edges<<<NEDGE / 8, 256, 0, stream>>>(ubf, tvb, user_indices, item_indices, out);
}
